// TemporalExtractor_49168785605458
// MI455X (gfx1250) — matrix operand order repaired and checked by reading, not compiled, not run on hardware
//
#include <hip/hip_runtime.h>
#include <hip/hip_bf16.h>

// ---------------------------------------------------------------------------
// Types for WMMA fragments (bf16 stored as raw u16 bits; bit_cast at the call)
// ---------------------------------------------------------------------------
typedef __attribute__((ext_vector_type(16))) __bf16          v16bf;
typedef __attribute__((ext_vector_type(8)))  float           v8f;
typedef __attribute__((ext_vector_type(8)))  unsigned short  v8us;
typedef __attribute__((ext_vector_type(16))) unsigned short  v16us;

#define DEV __device__ __forceinline__

DEV unsigned short f2bf(float f) {
    unsigned u = __builtin_bit_cast(unsigned, f);
    unsigned r = u + 0x7FFFu + ((u >> 16) & 1u);   // round-to-nearest-even
    return (unsigned short)(r >> 16);
}

DEV v16us comb(v8us a, v8us b) {
    v16us r;
#pragma unroll
    for (int e = 0; e < 8; ++e) { r[e] = a[e]; r[e + 8] = b[e]; }
    return r;
}

DEV v8f wmma_bf16(v16us a, v16us b, v8f c) {
    return __builtin_amdgcn_wmma_f32_16x16x32_bf16(
        false, __builtin_bit_cast(v16bf, a),
        false, __builtin_bit_cast(v16bf, b),
        (short)0, c, false, false);
}

DEV float wave_sum(float v) {
#pragma unroll
    for (int off = 16; off >= 1; off >>= 1) v += __shfl_xor(v, off, 32);
    return v;
}

// Problem dims
#define BB   32
#define TT   1024
#define FF   64
#define HH   768
#define CH   256
#define GG   4
#define DD   192
#define NN   (BB * TT)       // 32768 rows

// ===========================================================================
// Kernel 1: multi-scale gated conv + concat + LayerNorm  (fp32 VALU; L2-bound)
// One block handles 4 consecutive time steps of one batch row.
// ===========================================================================
__global__ void conv_front_kernel(
    const float* __restrict__ x,
    const float* __restrict__ w3f, const float* __restrict__ b3f,
    const float* __restrict__ w3g, const float* __restrict__ b3g,
    const float* __restrict__ w5f, const float* __restrict__ b5f,
    const float* __restrict__ w5g, const float* __restrict__ b5g,
    const float* __restrict__ w7f, const float* __restrict__ b7f,
    const float* __restrict__ w7g, const float* __restrict__ b7g,
    const float* __restrict__ lng, const float* __restrict__ lnb,
    float* __restrict__ h, unsigned short* __restrict__ hb)
{
    __shared__ float win[10 * FF];      // rows t0-3 .. t0+6
    __shared__ float rows[4 * HH];
    __shared__ float red[8], red2[8];

    const int tid  = threadIdx.x;
    const int b    = blockIdx.x / (TT / 4);
    const int t0   = (blockIdx.x % (TT / 4)) * 4;
    const int wid  = tid >> 5, lane = tid & 31;

    for (int i = tid; i < 10 * FF; i += 256) {
        int r = i / FF, f = i % FF;
        int t = t0 + r - 3;
        win[i] = (t >= 0 && t < TT) ? x[((size_t)b * TT + t) * FF + f] : 0.f;
    }
    __syncthreads();

#pragma unroll
    for (int e = 0; e < 3; ++e) {
        const int c = tid * 3 + e;
        const float *wf, *wg; float bfv, bgv; int k, cc;
        if (c < CH)           { k = 3; cc = c;          wf = w3f; wg = w3g; bfv = b3f[cc]; bgv = b3g[cc]; }
        else if (c < 2 * CH)  { k = 5; cc = c - CH;     wf = w5f; wg = w5g; bfv = b5f[cc]; bgv = b5g[cc]; }
        else                  { k = 7; cc = c - 2 * CH; wf = w7f; wg = w7g; bfv = b7f[cc]; bgv = b7g[cc]; }
        float af[4], ag[4];
#pragma unroll
        for (int t = 0; t < 4; ++t) { af[t] = bfv; ag[t] = bgv; }
        const int pad = k >> 1;
        for (int j = 0; j < k; ++j) {
            const int wb = (j - pad + 3) * FF;
            for (int f = 0; f < FF; ++f) {
                float wfv = wf[(j * FF + f) * CH + cc];
                float wgv = wg[(j * FF + f) * CH + cc];
#pragma unroll
                for (int t = 0; t < 4; ++t) {
                    float xv = win[wb + t * FF + f];
                    af[t] = fmaf(xv, wfv, af[t]);
                    ag[t] = fmaf(xv, wgv, ag[t]);
                }
            }
        }
#pragma unroll
        for (int t = 0; t < 4; ++t)
            rows[t * HH + c] = fmaxf(af[t], 0.f) * (1.f / (1.f + __expf(-ag[t])));
    }
    __syncthreads();

    for (int t = 0; t < 4; ++t) {
        float s = 0.f, q = 0.f;
#pragma unroll
        for (int e = 0; e < 3; ++e) {
            float v = rows[t * HH + tid * 3 + e];
            s += v; q += v * v;
        }
        s = wave_sum(s); q = wave_sum(q);
        if (lane == 0) { red[wid] = s; red2[wid] = q; }
        __syncthreads();
        float S = 0.f, Q = 0.f;
#pragma unroll
        for (int w = 0; w < 8; ++w) { S += red[w]; Q += red2[w]; }
        float mu = S * (1.f / HH);
        float rs = rsqrtf(Q * (1.f / HH) - mu * mu + 1e-5f);
#pragma unroll
        for (int e = 0; e < 3; ++e) {
            int c = tid * 3 + e;
            float y = (rows[t * HH + c] - mu) * rs * lng[c] + lnb[c];
            size_t idx = ((size_t)b * TT + t0 + t) * HH + c;
            h[idx]  = y;
            hb[idx] = f2bf(y);
        }
        __syncthreads();
    }
}

// ===========================================================================
// Kernel 2: weight convert + transpose  (fp32 KxN -> bf16 NxK, B-fragment order)
// ===========================================================================
__global__ void wt_convert_kernel(const float* __restrict__ W,
                                  unsigned short* __restrict__ WT,
                                  int K, int N)
{
    int idx = blockIdx.x * blockDim.x + threadIdx.x;
    if (idx >= K * N) return;
    int k = idx / N, n = idx % N;
    WT[(size_t)n * K + k] = f2bf(W[idx]);
}

// ===========================================================================
// Kernel 3: bf16 WMMA GEMM.  C(MxN) = A(MxK,row-major) * B(KxN)
// B supplied transposed (BT: N x K).  Block tile 64x128, 8 waves (4M x 2N),
// each wave: one A-fragment reused over 4 WMMA column tiles per K-step.
// ===========================================================================
template <bool OUT_BF16>
__global__ __launch_bounds__(256, 2)
void gemm_bf16_kernel(const unsigned short* __restrict__ A, int lda,
                      const unsigned short* __restrict__ BT,
                      float* __restrict__ Cf,
                      unsigned short* __restrict__ Cb,
                      int ldc, int Kdim)
{
    __shared__ unsigned short As[64 * 32];
    __shared__ unsigned short Bs[128 * 32];

    const int tid  = threadIdx.x;
    const int wid  = tid >> 5, lane = tid & 31;
    const int lh   = lane & 15, hi = lane >> 4;
    const size_t mbase = (size_t)blockIdx.y * 64;
    const size_t nbase = (size_t)blockIdx.x * 128;

    v8f acc[4];
#pragma unroll
    for (int j = 0; j < 4; ++j)
#pragma unroll
        for (int e = 0; e < 8; ++e) acc[j][e] = 0.f;

    for (int kk = 0; kk < Kdim; kk += 32) {
        // stage A tile (64x32) and B tile (128 cols x 32 K) into LDS
        {
            int r = tid >> 2, ch = (tid & 3) * 8;
            *(v8us*)&As[r * 32 + ch] =
                *(const v8us*)&A[(mbase + r) * (size_t)lda + kk + ch];
        }
#pragma unroll
        for (int rep = 0; rep < 2; ++rep) {
            int i = tid + rep * 256;
            int r = i >> 2, ch = (i & 3) * 8;
            *(v8us*)&Bs[r * 32 + ch] =
                *(const v8us*)&BT[(nbase + r) * (size_t)Kdim + kk + ch];
        }
        if (kk + 32 < Kdim)   // hint the next A tile toward the caches
            __builtin_prefetch(&A[(mbase + (tid >> 2)) * (size_t)lda + kk + 32], 0, 1);
        __syncthreads();

        const int arow = (wid & 3) * 16 + lh;
        const int alo  = hi ? 8 : 0;
        v16us av = comb(*(const v8us*)&As[arow * 32 + alo],
                        *(const v8us*)&As[arow * 32 + alo + 16]);
#pragma unroll
        for (int j = 0; j < 4; ++j) {
            const int brow = (wid >> 2) * 64 + j * 16 + lh;
            const int blo  = hi ? 8 : 0;
            v16us bv = comb(*(const v8us*)&Bs[brow * 32 + blo],
                            *(const v8us*)&Bs[brow * 32 + blo + 16]);
            acc[j] = wmma_bf16(av, bv, acc[j]);
        }
        __syncthreads();
    }

#pragma unroll
    for (int j = 0; j < 4; ++j)
#pragma unroll
        for (int vv = 0; vv < 8; ++vv) {
            size_t r = mbase + (wid & 3) * 16 + vv + 8 * hi;
            size_t c = nbase + (wid >> 2) * 64 + j * 16 + lh;
            if constexpr (OUT_BF16) Cb[r * ldc + c] = f2bf(acc[j][vv]);
            else                    Cf[r * ldc + c] = acc[j][vv];
        }
}

// ===========================================================================
// Kernel 4: flash attention, one block = 4 waves = 64 Q rows of one (b,g).
// Q tile staged in LDS (A-fragments re-read per chunk -> no VGPR spills;
// O accumulator 96 VGPR/lane stays resident).  S = Q K^T via WMMA, online
// softmax with 16-lane shuffles, O += P V via WMMA with V transposed in LDS.
// ===========================================================================
__global__ __launch_bounds__(128, 1)
void attn_kernel(const unsigned short* __restrict__ q,
                 const unsigned short* __restrict__ k,
                 const unsigned short* __restrict__ v,
                 unsigned short* __restrict__ o)
{
    __shared__ unsigned short Qs[64 * DD];     // Q tile, row-major (24 KB)
    __shared__ unsigned short Ks[32 * DD];     // K chunk, row-major (12 KB)
    __shared__ unsigned short Vt[DD * 32];     // V chunk, transposed (12 KB)
    __shared__ unsigned short Ps[4 * 16 * 32]; // per-wave P tile (4 KB)

    const int tid  = threadIdx.x;
    const int wid  = tid >> 5, lane = tid & 31;
    const int lh   = lane & 15, hi = lane >> 4;
    const int g    = blockIdx.y, b = blockIdx.z;
    const int qrow0 = blockIdx.x * 64;          // block's first Q row
    const size_t base = ((size_t)b * TT) * HH + g * DD;
    const float scale = 0.0721687836487032f;    // 1/sqrt(192)

    // stage the whole Q tile into LDS once
#pragma unroll
    for (int rep = 0; rep < 12; ++rep) {
        int i  = tid + rep * 128;               // 1536 vec8 loads
        int r  = i / 24, c0 = (i % 24) * 8;
        *(v8us*)&Qs[r * DD + c0] =
            *(const v8us*)&q[base + (size_t)(qrow0 + r) * HH + c0];
    }

    float m8[8], l8[8];
    v8f Oacc[12];
#pragma unroll
    for (int e = 0; e < 8; ++e) { m8[e] = -1e30f; l8[e] = 0.f; }
#pragma unroll
    for (int c = 0; c < 12; ++c)
#pragma unroll
        for (int e = 0; e < 8; ++e) Oacc[c][e] = 0.f;

    __syncthreads();

    const int aqrow = wid * 16 + lh;            // this lane's Q row in Qs
    const int alo   = hi ? 8 : 0;

    for (int kt = 0; kt < TT / 32; ++kt) {
        // cooperative stage of K (row-major) and V (transposed) chunks
#pragma unroll
        for (int rep = 0; rep < 6; ++rep) {
            int i  = tid + rep * 128;           // 768 vec8 loads
            int r  = i / 24, c0 = (i % 24) * 8;
            size_t gp = base + ((size_t)kt * 32 + r) * HH + c0;
            *(v8us*)&Ks[r * DD + c0] = *(const v8us*)&k[gp];
            v8us vv = *(const v8us*)&v[gp];
#pragma unroll
            for (int e = 0; e < 8; ++e) Vt[(c0 + e) * 32 + r] = vv[e];
        }
        __syncthreads();

        // ---- S = Q * K^T (16 x 32) ----
        v8f s0, s1;
#pragma unroll
        for (int e = 0; e < 8; ++e) { s0[e] = 0.f; s1[e] = 0.f; }
#pragma unroll
        for (int kd = 0; kd < 6; ++kd) {
            v16us av = comb(*(const v8us*)&Qs[aqrow * DD + kd * 32 + alo],
                            *(const v8us*)&Qs[aqrow * DD + kd * 32 + alo + 16]);
            const int dlo = kd * 32 + (hi ? 8 : 0);
            v16us b0 = comb(*(const v8us*)&Ks[lh * DD + dlo],
                            *(const v8us*)&Ks[lh * DD + dlo + 16]);
            s0 = wmma_bf16(av, b0, s0);
            v16us b1 = comb(*(const v8us*)&Ks[(16 + lh) * DD + dlo],
                            *(const v8us*)&Ks[(16 + lh) * DD + dlo + 16]);
            s1 = wmma_bf16(av, b1, s1);
        }

        // ---- online softmax over the 32 new columns ----
#pragma unroll
        for (int e = 0; e < 8; ++e) {
            float a0 = s0[e] * scale, a1 = s1[e] * scale;
            float mv = fmaxf(a0, a1);
#pragma unroll
            for (int off = 8; off >= 1; off >>= 1)
                mv = fmaxf(mv, __shfl_xor(mv, off, 32));
            float mnew = fmaxf(m8[e], mv);
            float corr = __expf(m8[e] - mnew);
            float p0 = __expf(a0 - mnew), p1 = __expf(a1 - mnew);
            float rs = p0 + p1;
#pragma unroll
            for (int off = 8; off >= 1; off >>= 1)
                rs += __shfl_xor(rs, off, 32);
            l8[e] = l8[e] * corr + rs;
            m8[e] = mnew;
#pragma unroll
            for (int c = 0; c < 12; ++c) Oacc[c][e] *= corr;
            const int prow = e + 8 * hi;
            Ps[wid * 512 + prow * 32 + lh]      = f2bf(p0);
            Ps[wid * 512 + prow * 32 + 16 + lh] = f2bf(p1);
        }
        // wave-internal LDS RAW: lanes read other lanes' P stores next
        asm volatile("s_wait_dscnt 0x0" ::: "memory");

        // ---- O += P * V ----
        const int plo = hi ? 8 : 0;
        v16us pv = comb(*(const v8us*)&Ps[wid * 512 + lh * 32 + plo],
                        *(const v8us*)&Ps[wid * 512 + lh * 32 + plo + 16]);
#pragma unroll
        for (int c = 0; c < 12; ++c) {
            const int n = c * 16 + lh;
            v16us bv = comb(*(const v8us*)&Vt[n * 32 + (hi ? 8 : 0)],
                            *(const v8us*)&Vt[n * 32 + (hi ? 8 : 0) + 16]);
            Oacc[c] = wmma_bf16(pv, bv, Oacc[c]);
        }
        __syncthreads();   // before next chunk overwrites Ks/Vt
    }

    // epilogue: normalize and store bf16
#pragma unroll
    for (int e = 0; e < 8; ++e) {
        const float inv = 1.f / l8[e];
        const size_t r = qrow0 + wid * 16 + e + 8 * hi;
#pragma unroll
        for (int c = 0; c < 12; ++c)
            o[base + r * HH + c * 16 + lh] = f2bf(Oacc[c][e] * inv);
    }
}

// ===========================================================================
// Kernel 5: residual + bias + LayerNorm  (h = LN(h + psum + bp))
// ===========================================================================
__global__ void resln_kernel(float* __restrict__ h, unsigned short* __restrict__ hb,
                             const float* __restrict__ psum,
                             const float* __restrict__ bp,
                             const float* __restrict__ g, const float* __restrict__ bt)
{
    __shared__ float rows[HH];
    __shared__ float red[8], red2[8];
    const int tid = threadIdx.x, wid = tid >> 5, lane = tid & 31;
    const size_t row = blockIdx.x;

    float s = 0.f, q = 0.f;
#pragma unroll
    for (int e = 0; e < 3; ++e) {
        int c = tid * 3 + e;
        float v = h[row * HH + c] + psum[row * HH + c] + bp[c];
        rows[c] = v; s += v; q += v * v;
    }
    s = wave_sum(s); q = wave_sum(q);
    if (lane == 0) { red[wid] = s; red2[wid] = q; }
    __syncthreads();
    float S = 0.f, Q = 0.f;
#pragma unroll
    for (int w = 0; w < 8; ++w) { S += red[w]; Q += red2[w]; }
    float mu = S * (1.f / HH);
    float rs = rsqrtf(Q * (1.f / HH) - mu * mu + 1e-5f);
#pragma unroll
    for (int e = 0; e < 3; ++e) {
        int c = tid * 3 + e;
        float y = (rows[c] - mu) * rs * g[c] + bt[c];
        h[row * HH + c]  = y;
        hb[row * HH + c] = f2bf(y);
    }
}

// ===========================================================================
// Kernel 6: mean + population-std pooling over T
// ===========================================================================
__global__ void pool_kernel(const float* __restrict__ h, float* __restrict__ pooled)
{
    __shared__ float red[8], red2[8];
    const int tid = threadIdx.x, wid = tid >> 5, lane = tid & 31;
    const int c = blockIdx.x, b = blockIdx.y;
    float s = 0.f, q = 0.f;
    for (int t = tid; t < TT; t += 256) {
        float v = h[((size_t)b * TT + t) * HH + c];
        s += v; q += v * v;
    }
    s = wave_sum(s); q = wave_sum(q);
    if (lane == 0) { red[wid] = s; red2[wid] = q; }
    __syncthreads();
    if (tid == 0) {
        float S = 0.f, Q = 0.f;
#pragma unroll
        for (int w = 0; w < 8; ++w) { S += red[w]; Q += red2[w]; }
        float mu = S * (1.f / TT);
        float var = fmaxf(Q * (1.f / TT) - mu * mu, 0.f);
        pooled[(size_t)b * (2 * HH) + c]      = mu;
        pooled[(size_t)b * (2 * HH) + HH + c] = sqrtf(var);
    }
}

// ===========================================================================
// Kernel 7: final projection + LayerNorm + ReLU  (32 x 1536 @ 1536 x 768)
// ===========================================================================
__global__ void final_kernel(const float* __restrict__ pooled,
                             const float* __restrict__ W, const float* __restrict__ bias,
                             const float* __restrict__ g, const float* __restrict__ bt,
                             float* __restrict__ out)
{
    __shared__ float pr[2 * HH];
    __shared__ float red[8], red2[8];
    const int tid = threadIdx.x, wid = tid >> 5, lane = tid & 31;
    const int b = blockIdx.x;

    for (int i = tid; i < 2 * HH; i += 256) pr[i] = pooled[(size_t)b * (2 * HH) + i];
    __syncthreads();

    float s = 0.f, q = 0.f;
    float zl[3];
#pragma unroll
    for (int e = 0; e < 3; ++e) {
        int c = tid * 3 + e;
        float acc = bias[c];
        for (int j = 0; j < 2 * HH; ++j) acc = fmaf(pr[j], W[(size_t)j * HH + c], acc);
        zl[e] = acc; s += acc; q += acc * acc;
    }
    s = wave_sum(s); q = wave_sum(q);
    if (lane == 0) { red[wid] = s; red2[wid] = q; }
    __syncthreads();
    float S = 0.f, Q = 0.f;
#pragma unroll
    for (int w = 0; w < 8; ++w) { S += red[w]; Q += red2[w]; }
    float mu = S * (1.f / HH);
    float rs = rsqrtf(Q * (1.f / HH) - mu * mu + 1e-5f);
#pragma unroll
    for (int e = 0; e < 3; ++e) {
        int c = tid * 3 + e;
        out[(size_t)b * HH + c] = fmaxf((zl[e] - mu) * rs * g[c] + bt[c], 0.f);
    }
}

// ===========================================================================
// Launch
// ===========================================================================
extern "C" void kernel_launch(void* const* d_in, const int* in_sizes, int n_in,
                              void* d_out, int out_size, void* d_ws, size_t ws_size,
                              hipStream_t stream)
{
    const float* X    = (const float*)d_in[0];
    const float* w3f  = (const float*)d_in[1];  const float* b3f = (const float*)d_in[2];
    const float* w3g  = (const float*)d_in[3];  const float* b3g = (const float*)d_in[4];
    const float* w5f  = (const float*)d_in[5];  const float* b5f = (const float*)d_in[6];
    const float* w5g  = (const float*)d_in[7];  const float* b5g = (const float*)d_in[8];
    const float* w7f  = (const float*)d_in[9];  const float* b7f = (const float*)d_in[10];
    const float* w7g  = (const float*)d_in[11]; const float* b7g = (const float*)d_in[12];
    const float* clng = (const float*)d_in[13]; const float* clnb = (const float*)d_in[14];
    const float* gwq  = (const float*)d_in[15];
    const float* gwk  = (const float*)d_in[16];
    const float* gwv  = (const float*)d_in[17];
    const float* gwp  = (const float*)d_in[18];
    const float* gbp  = (const float*)d_in[19];
    const float* glg  = (const float*)d_in[20];
    const float* glb  = (const float*)d_in[21];
    const float* pw   = (const float*)d_in[22]; const float* pb  = (const float*)d_in[23];
    const float* plg  = (const float*)d_in[24]; const float* plb = (const float*)d_in[25];

    const size_t NH = (size_t)NN * HH;
    char* ws = (char*)d_ws;
    float*          h      = (float*)ws;                     ws += NH * 4;
    unsigned short* hb     = (unsigned short*)ws;            ws += NH * 2;
    unsigned short* qb     = (unsigned short*)ws;            ws += NH * 2;
    unsigned short* kb     = (unsigned short*)ws;            ws += NH * 2;
    unsigned short* vb     = (unsigned short*)ws;            ws += NH * 2;
    unsigned short* aob    = (unsigned short*)ws;            ws += NH * 2;
    float*          psum   = (float*)ws;                     ws += NH * 4;
    unsigned short* wT     = (unsigned short*)ws;            ws += (size_t)4 * HH * HH * 2;
    float*          pooled = (float*)ws;

    // 1) gated conv front end + concat + LN
    conv_front_kernel<<<NN / 4, 256, 0, stream>>>(
        X, w3f, b3f, w3g, b3g, w5f, b5f, w5g, b5g, w7f, b7f, w7g, b7g,
        clng, clnb, h, hb);

    const dim3 ggrid(HH / 128, NN / 64);
    const int  wcb = (HH * HH + 255) / 256;

    // 2) two GQA blocks
    for (int i = 0; i < 2; ++i) {
        const size_t woff = (size_t)i * HH * HH;
        wt_convert_kernel<<<wcb, 256, 0, stream>>>(gwq + woff, wT + 0 * (size_t)HH * HH, HH, HH);
        wt_convert_kernel<<<wcb, 256, 0, stream>>>(gwk + woff, wT + 1 * (size_t)HH * HH, HH, HH);
        wt_convert_kernel<<<wcb, 256, 0, stream>>>(gwv + woff, wT + 2 * (size_t)HH * HH, HH, HH);
        wt_convert_kernel<<<wcb, 256, 0, stream>>>(gwp + woff, wT + 3 * (size_t)HH * HH, HH, HH);

        gemm_bf16_kernel<true><<<ggrid, 256, 0, stream>>>(hb, HH, wT + 0 * (size_t)HH * HH, nullptr, qb, HH, HH);
        gemm_bf16_kernel<true><<<ggrid, 256, 0, stream>>>(hb, HH, wT + 1 * (size_t)HH * HH, nullptr, kb, HH, HH);
        gemm_bf16_kernel<true><<<ggrid, 256, 0, stream>>>(hb, HH, wT + 2 * (size_t)HH * HH, nullptr, vb, HH, HH);

        attn_kernel<<<dim3(TT / 64, GG, BB), 128, 0, stream>>>(qb, kb, vb, aob);

        gemm_bf16_kernel<false><<<ggrid, 256, 0, stream>>>(aob, HH, wT + 3 * (size_t)HH * HH, psum, nullptr, HH, HH);

        resln_kernel<<<NN, 256, 0, stream>>>(h, hb, psum, gbp + (size_t)i * HH,
                                             glg + (size_t)i * HH, glb + (size_t)i * HH);
    }

    // 3) mean/std pooling over T
    pool_kernel<<<dim3(HH, BB), 256, 0, stream>>>(h, pooled);

    // 4) final projection + LN + ReLU
    final_kernel<<<BB, 256, 0, stream>>>(pooled, pw, pb, plg, plb, (float*)d_out);
}
